// GCNModelARGA_21749714387358
// MI455X (gfx1250) — matrix operand order repaired and checked by reading, not compiled, not run on hardware
//
#include <hip/hip_runtime.h>
#include <hip/hip_bf16.h>
#include <math.h>

// ---------------------------------------------------------------------------
// GCN-VAE (ARGA) forward for MI455X / gfx1250 (wave32, WMMA).
// Dense GEMMs use v_wmma_f32_16x16x32_bf16 with f32 accumulation.
// Decoder z@z^T (rank-64, 8.6 GFLOP) gets a specialized 1x4 register-blocked
// kernel: A fragment loaded once, 4 accumulators, 8 WMMAs per wave.
// SpMM (COO edge list) uses L2-resident gather + global_atomic_add_f32.
// ---------------------------------------------------------------------------

typedef __attribute__((ext_vector_type(16))) __bf16 v16bf;
typedef __attribute__((ext_vector_type(8)))  __bf16 bf16x8;
typedef __attribute__((ext_vector_type(8)))  float  v8f;

#define NNODES 8192
#define NEDGES 524288
#define FIN    512
#define H1D    256
#define H2D    64

union FragU { struct { bf16x8 lo, hi; } p; v16bf v; };

// ---- f32 -> bf16 conversion (optionally fused ReLU) -----------------------
__global__ void cvt_bf16_kernel(const float* __restrict__ src,
                                __bf16* __restrict__ dst, int n, int relu) {
  int i = blockIdx.x * blockDim.x + threadIdx.x;
  if (i >= n) return;
  float v = src[i];
  if (relu) v = fmaxf(v, 0.0f);
  dst[i] = (__bf16)v;
}

// ---- generic bf16 WMMA GEMM: C[M,Nc] = A[M,K] * B[K,Nc], all row-major ----
// One 16x16 output tile per wave; block = 8 waves (256 threads).
__global__ __launch_bounds__(256) void gemm_bf16_wmma_kernel(
    const __bf16* __restrict__ A, const __bf16* __restrict__ B,
    float* __restrict__ C, int M, int K, int Nc) {
  const int lane = threadIdx.x;               // 0..31
  const int wave = threadIdx.y;               // 0..7
  const int nTilesN = Nc >> 4;
  const int numTiles = (M >> 4) * nTilesN;
  const int t = blockIdx.x * 8 + wave;
  if (t >= numTiles) return;                  // wave-uniform: EXEC stays all-1s
  const int tm = t / nTilesN;
  const int tn = t - tm * nTilesN;

  const int hi   = lane >> 4;                 // 0 or 1 (lane half)
  const int lo16 = lane & 15;
  const int kseg = hi << 3;                   // A: K-chunk base 0 or 8
  const int bk   = lo16 + (hi << 4);          // B: K row within 32-chunk

  const __bf16* Arow = A + (size_t)((tm << 4) + lo16) * K;
  const __bf16* Bcol = B + (size_t)(tn << 4);

  v8f acc = {};
  for (int k0 = 0; k0 < K; k0 += 32) {
    FragU a, b;
    a.p.lo = *(const bf16x8*)(Arow + k0 + kseg);
    a.p.hi = *(const bf16x8*)(Arow + k0 + kseg + 16);
    const __bf16* Bc = Bcol + lo16;
#pragma unroll
    for (int s = 0; s < 16; ++s)
      b.v[s] = Bc[(size_t)(k0 + kseg + (s & 7) + ((s >> 3) << 4)) * Nc];
    if (k0 + 32 < K) {
      __builtin_prefetch(Arow + k0 + 32 + kseg, 0, 1);          // global_prefetch_b8
      __builtin_prefetch(Bcol + (size_t)(k0 + 32 + bk) * Nc, 0, 1);
    }
    // D = A x B + C   (f32 accumulate)
    acc = __builtin_amdgcn_wmma_f32_16x16x32_bf16(
        /*neg_a=*/false, a.v, /*neg_b=*/false, b.v,
        /*c_mod=*/(short)0, acc, /*reuse_a=*/false, /*reuse_b=*/false);
  }
  // C/D layout: VGPR r -> row (r + 8*hi), col = lo16
  float* Crow = C + (size_t)((tm << 4) + (hi << 3)) * Nc + (tn << 4) + lo16;
#pragma unroll
  for (int r = 0; r < 8; ++r) Crow[(size_t)r * Nc] = acc[r];
}

// ---- specialized decoder: C[8192,8192] = Z[8192,64] * ZT[64,8192] ---------
// 1x4 register blocking: A fragment (both K-halves) loaded once per wave,
// reused across 4 column tiles -> 8 independent WMMAs, 4 accumulators.
// Grid: (16, 512); block (32, 8). Wave covers 16 rows x 64 cols.
__global__ __launch_bounds__(256) void zzT_wmma_kernel(
    const __bf16* __restrict__ Z, const __bf16* __restrict__ ZT,
    float* __restrict__ C) {
  const int lane = threadIdx.x;
  const int wave = threadIdx.y;
  const int tm  = blockIdx.y;                       // row tile 0..511
  const int tn0 = (blockIdx.x * 8 + wave) * 4;      // first of 4 col tiles

  const int hi   = lane >> 4;
  const int lo16 = lane & 15;
  const int kseg = hi << 3;
  const int bk   = lo16 + (hi << 4);

  // A fragments: row (tm*16 + lo16), K-halves [0,32) and [32,64)
  const __bf16* Arow = Z + (size_t)((tm << 4) + lo16) * H2D;
  FragU a0, a1;
  a0.p.lo = *(const bf16x8*)(Arow + kseg);
  a0.p.hi = *(const bf16x8*)(Arow + kseg + 16);
  a1.p.lo = *(const bf16x8*)(Arow + 32 + kseg);
  a1.p.hi = *(const bf16x8*)(Arow + 32 + kseg + 16);

  const __bf16* B0 = ZT + (size_t)bk * NNODES;         // k = bk
  const __bf16* B1 = ZT + (size_t)(bk + 32) * NNODES;  // k = bk + 32

  v8f acc[4] = {};
#pragma unroll
  for (int j = 0; j < 4; ++j) {
    const int nb = (tn0 + j) << 4;
    FragU b0, b1;
    const __bf16* Bn = Z + (size_t)(nb + lo16) * H2D;
    b0.p.lo = *(const bf16x8*)(Bn + kseg);
    b0.p.hi = *(const bf16x8*)(Bn + kseg + 16);
    b1.p.lo = *(const bf16x8*)(Bn + 32 + kseg);
    b1.p.hi = *(const bf16x8*)(Bn + 32 + kseg + 16);
    acc[j] = __builtin_amdgcn_wmma_f32_16x16x32_bf16(
        false, a0.v, false, b0.v, (short)0, acc[j], false, false);
    acc[j] = __builtin_amdgcn_wmma_f32_16x16x32_bf16(
        false, a1.v, false, b1.v, (short)0, acc[j], false, false);
  }

  float* Crow = C + (size_t)((tm << 4) + (hi << 3)) * NNODES + lo16;
#pragma unroll
  for (int j = 0; j < 4; ++j) {
    const size_t nb = (size_t)((tn0 + j) << 4);
#pragma unroll
    for (int r = 0; r < 8; ++r)
      Crow[(size_t)r * NNODES + nb] = acc[j][r];
  }
}

// ---- COO SpMM: out[dst] += w * support[src], feature-vectorized by 4 ------
__global__ void spmm_atomic_kernel(const float* __restrict__ sup,
                                   const int* __restrict__ esrc,
                                   const int* __restrict__ edst,
                                   const float* __restrict__ ew,
                                   float* __restrict__ out,
                                   int dq /* d/4 */, int d, long long total) {
  long long i = (long long)blockIdx.x * blockDim.x + threadIdx.x;
  if (i >= total) return;
  int e = (int)(i / dq);
  int c = (int)(i - (long long)e * dq) << 2;
  int s = esrc[e], t = edst[e];
  float w = ew[e];
  const float4 v = *(const float4*)(sup + (size_t)s * d + c);
  float* o = out + (size_t)t * d + c;
  atomicAdd(o + 0, w * v.x);
  atomicAdd(o + 1, w * v.y);
  atomicAdd(o + 2, w * v.z);
  atomicAdd(o + 3, w * v.w);
}

// ---- reparameterize: z = eps*exp(logvar)+mu ; also emit bf16 copy ---------
__global__ void zfake_kernel(const float* __restrict__ eps,
                             const float* __restrict__ mu,
                             const float* __restrict__ logvar,
                             float* __restrict__ zf,
                             __bf16* __restrict__ zb, int n) {
  int i = blockIdx.x * blockDim.x + threadIdx.x;
  if (i >= n) return;
  float z = eps[i] * __expf(logvar[i]) + mu[i];
  zf[i] = z;
  zb[i] = (__bf16)z;
}

// ---- bf16 transpose: zbT[k, n] = zb[n, k]  (64 x 8192 <- 8192 x 64) -------
__global__ void transpose_bf16_kernel(const __bf16* __restrict__ zb,
                                      __bf16* __restrict__ zbT, int n) {
  int i = blockIdx.x * blockDim.x + threadIdx.x;
  if (i >= n) return;
  int r = i >> 6;       // node
  int k = i & 63;       // feature
  zbT[(size_t)k * NNODES + r] = zb[i];
}

// ---- discriminator MLP, one block (256 thr) per row -----------------------
__global__ __launch_bounds__(256) void disc_kernel(
    const float* __restrict__ z,
    const float* __restrict__ D1w, const float* __restrict__ D1b,
    const float* __restrict__ D2w, const float* __restrict__ D2b,
    const float* __restrict__ D3w, const float* __restrict__ D3b,
    float* __restrict__ out) {
  __shared__ float zrow[H2D];
  __shared__ float h1[H1D];
  __shared__ float h2[H2D];
  const int row = blockIdx.x;
  const int t = threadIdx.x;
  if (t < H2D) zrow[t] = z[(size_t)row * H2D + t];
  __syncthreads();
  // h1 = relu(z @ D1w + b1) : D1w is [64,256]
  float a = D1b[t];
#pragma unroll 8
  for (int k = 0; k < H2D; ++k) a += zrow[k] * D1w[k * H1D + t];
  h1[t] = fmaxf(a, 0.0f);
  __syncthreads();
  // h2 = relu(h1 @ D2w + b2) : D2w is [256,64]
  if (t < H2D) {
    float b = D2b[t];
#pragma unroll 8
    for (int k = 0; k < H1D; ++k) b += h1[k] * D2w[k * H2D + t];
    h2[t] = fmaxf(b, 0.0f);
  }
  __syncthreads();
  // out = h2 @ D3w + b3 : D3w is [64,1]
  if (t == 0) {
    float c = D3b[0];
#pragma unroll 8
    for (int k = 0; k < H2D; ++k) c += h2[k] * D3w[k];
    out[row] = c;
  }
}

// ---------------------------------------------------------------------------
extern "C" void kernel_launch(void* const* d_in, const int* in_sizes, int n_in,
                              void* d_out, int out_size, void* d_ws, size_t ws_size,
                              hipStream_t stream) {
  (void)in_sizes; (void)n_in; (void)out_size; (void)ws_size;

  const float* x     = (const float*)d_in[0];
  const int*   esrc  = (const int*)  d_in[1];
  const int*   edst  = (const int*)  d_in[2];
  const float* ew    = (const float*)d_in[3];
  const float* eps   = (const float*)d_in[4];
  const float* zreal = (const float*)d_in[5];
  const float* W1    = (const float*)d_in[6];
  const float* W2    = (const float*)d_in[7];
  const float* W3    = (const float*)d_in[8];
  const float* D1w   = (const float*)d_in[9];
  const float* D1b   = (const float*)d_in[10];
  const float* D2w   = (const float*)d_in[11];
  const float* D2b   = (const float*)d_in[12];
  const float* D3w   = (const float*)d_in[13];
  const float* D3b   = (const float*)d_in[14];

  const int N = NNODES, E = NEDGES, F = FIN, H1 = H1D, H2 = H2D;

  // ---- workspace carve-out (~38 MB) ----
  char* ws = (char*)d_ws;
  size_t off = 0;
  auto carve = [&](size_t bytes) {
    void* p = ws + off;
    off += (bytes + 255) & ~(size_t)255;
    return p;
  };
  float*  support1 = (float*) carve((size_t)N * H1 * 4);   // 8 MB
  float*  hidden1  = (float*) carve((size_t)N * H1 * 4);   // 8 MB (atomic accum)
  float*  support2 = (float*) carve((size_t)N * H2 * 4);   // 2 MB
  float*  support3 = (float*) carve((size_t)N * H2 * 4);   // 2 MB
  float*  zfake    = (float*) carve((size_t)N * H2 * 4);   // 2 MB
  __bf16* xb       = (__bf16*)carve((size_t)N * F  * 2);   // 8 MB
  __bf16* w1b      = (__bf16*)carve((size_t)F * H1 * 2);
  __bf16* w2b      = (__bf16*)carve((size_t)H1 * H2 * 2);
  __bf16* w3b      = (__bf16*)carve((size_t)H1 * H2 * 2);
  __bf16* h1b      = (__bf16*)carve((size_t)N * H1 * 2);   // 4 MB
  __bf16* zb       = (__bf16*)carve((size_t)N * H2 * 2);   // 1 MB
  __bf16* zbT      = (__bf16*)carve((size_t)N * H2 * 2);   // 1 MB

  // ---- output carve-out (tuple order: adj, dis_real, dis_fake, mu, logvar)
  float* out     = (float*)d_out;
  float* adj     = out;
  float* disr    = adj + (size_t)N * N;
  float* disf    = disr + N;
  float* mu      = disf + N;
  float* logvar  = mu + (size_t)N * H2;

  const dim3 gblk(32, 8);  // 8 waves of 32 (wave32)

  // zero the atomic accumulators (graph-capture-legal memsets)
  hipMemsetAsync(hidden1, 0, (size_t)N * H1 * 4, stream);
  hipMemsetAsync(mu,      0, (size_t)N * H2 * 4 * 2, stream);  // mu + logvar contiguous

  // bf16 operand staging
  cvt_bf16_kernel<<<(N * F + 255) / 256, 256, 0, stream>>>(x,  xb,  N * F,  0);
  cvt_bf16_kernel<<<(F * H1 + 255) / 256, 256, 0, stream>>>(W1, w1b, F * H1, 0);
  cvt_bf16_kernel<<<(H1 * H2 + 255) / 256, 256, 0, stream>>>(W2, w2b, H1 * H2, 0);
  cvt_bf16_kernel<<<(H1 * H2 + 255) / 256, 256, 0, stream>>>(W3, w3b, H1 * H2, 0);

  // GCN layer 1: support1 = x @ W1   [8192,512]x[512,256]
  {
    int tiles = (N / 16) * (H1 / 16);            // 8192
    gemm_bf16_wmma_kernel<<<tiles / 8, gblk, 0, stream>>>(xb, w1b, support1, N, F, H1);
  }
  // hidden1 = segsum(edge_w * support1[src]) ; relu fused into bf16 convert
  {
    long long total = (long long)E * (H1 / 4);
    spmm_atomic_kernel<<<(unsigned)((total + 255) / 256), 256, 0, stream>>>(
        support1, esrc, edst, ew, hidden1, H1 / 4, H1, total);
  }
  cvt_bf16_kernel<<<(N * H1 + 255) / 256, 256, 0, stream>>>(hidden1, h1b, N * H1, 1);

  // GCN layer 2/3: support2 = h1 @ W2 ; support3 = h1 @ W3   [8192,256]x[256,64]
  {
    int tiles = (N / 16) * (H2 / 16);            // 2048
    gemm_bf16_wmma_kernel<<<tiles / 8, gblk, 0, stream>>>(h1b, w2b, support2, N, H1, H2);
    gemm_bf16_wmma_kernel<<<tiles / 8, gblk, 0, stream>>>(h1b, w3b, support3, N, H1, H2);
  }
  // mu / logvar via SpMM (straight into d_out)
  {
    long long total = (long long)E * (H2 / 4);
    spmm_atomic_kernel<<<(unsigned)((total + 255) / 256), 256, 0, stream>>>(
        support2, esrc, edst, ew, mu, H2 / 4, H2, total);
    spmm_atomic_kernel<<<(unsigned)((total + 255) / 256), 256, 0, stream>>>(
        support3, esrc, edst, ew, logvar, H2 / 4, H2, total);
  }

  // z_fake = eps * exp(logvar) + mu ; bf16 copy + transpose for decoder
  zfake_kernel<<<(N * H2 + 255) / 256, 256, 0, stream>>>(eps, mu, logvar, zfake, zb, N * H2);
  transpose_bf16_kernel<<<(N * H2 + 255) / 256, 256, 0, stream>>>(zb, zbT, N * H2);

  // adj_rec = z_fake @ z_fake^T : specialized 1x4 register-blocked decoder
  // grid (16, 512): 16 * 8 waves * 4 tiles = 512 col tiles per row band
  zzT_wmma_kernel<<<dim3(16, 512), gblk, 0, stream>>>(zb, zbT, adj);

  // discriminators (exact f32, per-row MLP through LDS)
  disc_kernel<<<N, 256, 0, stream>>>(zreal, D1w, D1b, D2w, D2b, D3w, D3b, disr);
  disc_kernel<<<N, 256, 0, stream>>>(zfake, D1w, D1b, D2w, D2b, D3w, D3b, disf);
}
